// SoftSuperpixelNeighborhoodAttention_56126632624240
// MI455X (gfx1250) — matrix operand order repaired and checked by reading, not compiled, not run on hardware
//
#include <hip/hip_runtime.h>
#include <hip/hip_bf16.h>
#include <math.h>

typedef __attribute__((ext_vector_type(16))) __bf16 v16bf;
typedef __attribute__((ext_vector_type(8)))  float  v8f;

#define HH      192
#define WW      192
#define NPIX    (HH * WW)      // 36864
#define KW      7
#define QK_SCALE 0.25f         // head_dim^-0.5, head_dim = 16

// ---------------------------------------------------------------------------
// Kernel 1: fused Q/K/V projection.
//   x:   [64, NPIX]  (NCHW, so xf[p][c] = x[c*NPIX + p])
//   w:   [64, 64]    row-major (in, out)
//   qkv: [NPIX, 192] row-major, cols 0-63 = q, 64-127 = k, 128-191 = v
// One wave handles one 16-pixel M-tile; bf16 WMMA 16x16x32, f32 accumulate.
// ---------------------------------------------------------------------------
__global__ __launch_bounds__(256)
void qkv_proj_kernel(const float* __restrict__ x,
                     const float* __restrict__ wq,
                     const float* __restrict__ wk,
                     const float* __restrict__ wv,
                     float* __restrict__ qkv)
{
    __shared__ float lds[8][16][17];
    const int wave = threadIdx.x >> 5;
    const int lane = threadIdx.x & 31;
    const int p0   = (blockIdx.x * 8 + wave) * 16;
    const int m    = lane & 15;
    const int koff = (lane >> 4) << 4;      // A-layout: lanes 16-31 hold K+16
    const int p    = p0 + m;

    // A tile (16 pixels x 64 channels) as two 16x32 bf16 operands
    v16bf a0, a1;
#pragma unroll
    for (int e = 0; e < 16; ++e) {
        const int k = koff + e;   // koff+e
        a0[e] = (__bf16)x[k * NPIX + p];
        a1[e] = (__bf16)x[(k + 32) * NPIX + p];
    }

    const int n  = lane & 15;
    const int kb = (lane >> 4) << 4;        // B-layout: lanes 16-31 hold K+16

#pragma unroll
    for (int wsel = 0; wsel < 3; ++wsel) {
        const float* __restrict__ wm = (wsel == 0) ? wq : ((wsel == 1) ? wk : wv);
#pragma unroll
        for (int nt = 0; nt < 4; ++nt) {
            const int n0 = nt * 16;
            v16bf b0, b1;
#pragma unroll
            for (int e = 0; e < 16; ++e) {
                b0[e] = (__bf16)wm[(kb + e) * 64 + n0 + n];
                b1[e] = (__bf16)wm[(kb + e + 32) * 64 + n0 + n];
            }
            v8f c = {};
            c = __builtin_amdgcn_wmma_f32_16x16x32_bf16(false, a0, false, b0,
                                                        (short)0, c, false, false);
            c = __builtin_amdgcn_wmma_f32_16x16x32_bf16(false, a1, false, b1,
                                                        (short)0, c, false, false);
            // stage D tile as [m][n] in LDS, then write coalesced rows
#pragma unroll
            for (int r = 0; r < 8; ++r)
                lds[wave][r + ((lane >> 4) << 3)][n] = c[r];
            asm volatile("s_wait_dscnt 0" ::: "memory");
#pragma unroll
            for (int it = 0; it < 8; ++it) {
                const int mr = (lane >> 4) + (it << 1);
                qkv[(p0 + mr) * 192 + wsel * 64 + n0 + (lane & 15)] =
                    lds[wave][mr][lane & 15];
            }
            asm volatile("s_wait_dscnt 0" ::: "memory");
        }
    }
}

// ---------------------------------------------------------------------------
// Kernel 2: soft-superpixel neighborhood attention, online softmax.
// One thread = one (pixel, head). Single pass over the 49 neighbors:
// gathers k, v, and the sims mask exactly once each.
//   softmax over t_o = SCALE * (q.k_o) * m_o;  out = sum attn_o * m_o * v_o
// ---------------------------------------------------------------------------
__global__ __launch_bounds__(256)
void attn_kernel(const float* __restrict__ qkv,
                 const float* __restrict__ sims,   // [HH, WW, 12, 12]
                 float* __restrict__ aout)         // [NPIX, 64]
{
    const int gid = blockIdx.x * 256 + threadIdx.x;   // [0, NPIX*4)
    const int h   = gid & 3;
    const int p   = gid >> 2;
    const int y   = p / WW;
    const int x   = p - y * WW;
    const int sy  = min(max(y - 3, 0), HH - KW);
    const int sx  = min(max(x - 3, 0), WW - KW);
    const int si  = y >> 4;                 // (y*12)/192
    const int sj  = x >> 4;

    float q[16];
    {
        const float4* q4 = (const float4*)(qkv + p * 192 + h * 16);
#pragma unroll
        for (int i = 0; i < 4; ++i) {
            const float4 t = q4[i];
            q[4*i+0] = t.x; q[4*i+1] = t.y; q[4*i+2] = t.z; q[4*i+3] = t.w;
        }
    }

    float mx = -1e30f, ssum = 0.0f;
    float acc[16];
#pragma unroll
    for (int d = 0; d < 16; ++d) acc[d] = 0.0f;

    for (int ky = 0; ky < KW; ++ky) {
        const int ny = sy + ky;
        for (int kx = 0; kx < KW; ++kx) {
            const int nx  = sx + kx;
            const int np  = ny * WW + nx;
            const float* base = qkv + np * 192 + h * 16;
            const float4* k4 = (const float4*)(base + 64);
            const float4* v4 = (const float4*)(base + 128);
            float kk[16], vv[16];
#pragma unroll
            for (int i = 0; i < 4; ++i) {
                const float4 t = k4[i];
                kk[4*i+0] = t.x; kk[4*i+1] = t.y; kk[4*i+2] = t.z; kk[4*i+3] = t.w;
                const float4 u = v4[i];
                vv[4*i+0] = u.x; vv[4*i+1] = u.y; vv[4*i+2] = u.z; vv[4*i+3] = u.w;
            }
            float dot = 0.0f;
#pragma unroll
            for (int d = 0; d < 16; ++d) dot = fmaf(q[d], kk[d], dot);

            const float mval = sims[np * 144 + si * 12 + sj];
            const float val  = QK_SCALE * dot * mval;

            const float mnew = fmaxf(mx, val);
            const float cf   = __expf(mx - mnew);
            const float e    = __expf(val - mnew);
            ssum = ssum * cf + e;
            const float wgt = e * mval;       // post-softmax mask folded in
#pragma unroll
            for (int d = 0; d < 16; ++d) acc[d] = acc[d] * cf + wgt * vv[d];
            mx = mnew;
        }
    }

    const float inv = 1.0f / ssum;
    float4* o4 = (float4*)(aout + p * 64 + h * 16);
#pragma unroll
    for (int i = 0; i < 4; ++i) {
        float4 t;
        t.x = acc[4*i+0] * inv; t.y = acc[4*i+1] * inv;
        t.z = acc[4*i+2] * inv; t.w = acc[4*i+3] * inv;
        o4[i] = t;
    }
}

// ---------------------------------------------------------------------------
// Kernel 3: output projection + transpose back to NCHW.
//   ain: [NPIX, 64], wp: [64, 64], out: [64, NPIX]
// D tile is transposed through LDS so the NCHW store is 64B-coalesced.
// ---------------------------------------------------------------------------
__global__ __launch_bounds__(256)
void out_proj_kernel(const float* __restrict__ ain,
                     const float* __restrict__ wp,
                     float* __restrict__ out)
{
    __shared__ float lds[8][16][17];
    const int wave = threadIdx.x >> 5;
    const int lane = threadIdx.x & 31;
    const int p0   = (blockIdx.x * 8 + wave) * 16;
    const int m    = lane & 15;
    const int koff = (lane >> 4) << 4;

    v16bf a0, a1;
#pragma unroll
    for (int e = 0; e < 16; ++e) {
        const int k = koff + e;
        a0[e] = (__bf16)ain[(p0 + m) * 64 + k];
        a1[e] = (__bf16)ain[(p0 + m) * 64 + k + 32];
    }

    const int n  = lane & 15;
    const int kb = (lane >> 4) << 4;
#pragma unroll
    for (int nt = 0; nt < 4; ++nt) {
        const int n0 = nt * 16;
        v16bf b0, b1;
#pragma unroll
        for (int e = 0; e < 16; ++e) {
            b0[e] = (__bf16)wp[(kb + e) * 64 + n0 + n];
            b1[e] = (__bf16)wp[(kb + e + 32) * 64 + n0 + n];
        }
        v8f c = {};
        c = __builtin_amdgcn_wmma_f32_16x16x32_bf16(false, a0, false, b0,
                                                    (short)0, c, false, false);
        c = __builtin_amdgcn_wmma_f32_16x16x32_bf16(false, a1, false, b1,
                                                    (short)0, c, false, false);
        // stage transposed: lds[n][m]
#pragma unroll
        for (int r = 0; r < 8; ++r)
            lds[wave][n][r + ((lane >> 4) << 3)] = c[r];
        asm volatile("s_wait_dscnt 0" ::: "memory");
#pragma unroll
        for (int it = 0; it < 8; ++it) {
            const int nr = (lane >> 4) + (it << 1);
            out[(n0 + nr) * NPIX + p0 + (lane & 15)] = lds[wave][nr][lane & 15];
        }
        asm volatile("s_wait_dscnt 0" ::: "memory");
    }
}

// ---------------------------------------------------------------------------
extern "C" void kernel_launch(void* const* d_in, const int* in_sizes, int n_in,
                              void* d_out, int out_size, void* d_ws, size_t ws_size,
                              hipStream_t stream)
{
    const float* x    = (const float*)d_in[0];   // [1,64,192,192]
    const float* sims = (const float*)d_in[1];   // [1,192,192,12,12]
    const float* wq   = (const float*)d_in[2];
    const float* wk   = (const float*)d_in[3];
    const float* wv   = (const float*)d_in[4];
    const float* wp   = (const float*)d_in[5];
    float* out = (float*)d_out;                  // [1,64,192,192]

    float* qkv  = (float*)d_ws;                  // NPIX*192 f32 = 28.3 MB
    float* aout = qkv + (size_t)NPIX * 192;      // NPIX*64  f32 =  9.4 MB

    // 2304 M-tiles, 8 waves (tiles) per 256-thread block
    qkv_proj_kernel<<<NPIX / 16 / 8, 256, 0, stream>>>(x, wq, wk, wv, qkv);
    attn_kernel<<<(NPIX * 4) / 256, 256, 0, stream>>>(qkv, sims, aout);
    out_proj_kernel<<<NPIX / 16 / 8, 256, 0, stream>>>(aout, wp, out);
}
